// NSGEncoder_27633819583229
// MI455X (gfx1250) — matrix operand order repaired and checked by reading, not compiled, not run on hardware
//
#include <hip/hip_runtime.h>
#include <cstdint>
#include <cmath>

typedef __bf16 bf16;
typedef __attribute__((ext_vector_type(16))) __bf16 v16bf;
typedef __attribute__((ext_vector_type(8)))  __bf16 v8bf;
typedef __attribute__((ext_vector_type(8)))  float  v8f;

#define D_   512
#define MT_  64
#define B_   1024
#define H_   8
#define HD_  64
#define FF_  2048
#define NTOK 65536   // B_*MT_
#define NOBJ 8192
#define NTRI 10240

__device__ __forceinline__ float gelu_exact(float x) {
  return 0.5f * x * (1.0f + erff(x * 0.70710678118654752f));
}

// ---------------------------------------------------------------- utilities
__global__ void zero_f32(float* __restrict__ p, long long n) {
  long long i = (long long)blockIdx.x * blockDim.x + threadIdx.x;
  long long st = (long long)gridDim.x * blockDim.x;
  for (; i < n; i += st) p[i] = 0.0f;
}

__global__ void zero_u8(unsigned char* __restrict__ p, long long n) {
  long long i = (long long)blockIdx.x * blockDim.x + threadIdx.x;
  long long st = (long long)gridDim.x * blockDim.x;
  for (; i < n; i += st) p[i] = 0;
}

// W (K x N, f32, row-major) -> Wt (N x K, bf16) so WMMA B-fragments read contiguous K
__global__ void wt_convert(const float* __restrict__ W, bf16* __restrict__ Wt,
                           int K, int N) {
  long long n = (long long)K * N;
  long long i = (long long)blockIdx.x * blockDim.x + threadIdx.x;
  long long st = (long long)gridDim.x * blockDim.x;
  for (; i < n; i += st) {
    int k = (int)(i / N);
    int c = (int)(i % N);
    Wt[(long long)c * K + k] = (bf16)W[i];
  }
}

// t = boxes @ bw1 + bb1   (K=4 contraction, done scalar)
__global__ void boxlin(const float* __restrict__ boxes, const float* __restrict__ w,
                       const float* __restrict__ b, float* __restrict__ out) {
  int i = blockIdx.x * 256 + threadIdx.x;   // NOBJ * D_
  int row = i >> 9, d = i & 511;
  const float* bx = boxes + row * 4;
  out[i] = b[d] + bx[0]*w[d] + bx[1]*w[512+d] + bx[2]*w[1024+d] + bx[3]*w[1536+d];
}

// LayerNorm over rows of 512; one wave32 per row; optional exact-gelu; bf16 out
__global__ __launch_bounds__(256)
void ln_rows(const float* __restrict__ x, const float* __restrict__ g,
             const float* __restrict__ bt, bf16* __restrict__ out, int do_gelu) {
  int row  = blockIdx.x * 8 + (threadIdx.x >> 5);
  int lane = threadIdx.x & 31;
  const float* xr = x + (long long)row * D_;
  int base = lane * 16;
  float v[16]; float s = 0.0f, q = 0.0f;
#pragma unroll
  for (int i = 0; i < 16; ++i) { float t = xr[base + i]; v[i] = t; s += t; q += t * t; }
#pragma unroll
  for (int o = 16; o > 0; o >>= 1) { s += __shfl_xor(s, o, 32); q += __shfl_xor(q, o, 32); }
  float mean = s * (1.0f / D_);
  float var  = q * (1.0f / D_) - mean * mean;
  float rstd = rsqrtf(var + 1e-5f);
  bf16* orow = out + (long long)row * D_;
#pragma unroll
  for (int i = 0; i < 16; ++i) {
    int d = base + i;
    float hv = (v[i] - mean) * rstd * g[d] + bt[d];
    if (do_gelu) hv = gelu_exact(hv);
    orow[d] = (bf16)hv;
  }
}

// obj_tok[i,:] += obj_emb[objs[i],:]
__global__ void addemb(float* __restrict__ tok, const float* __restrict__ emb,
                       const int* __restrict__ objs) {
  int i = blockIdx.x * 256 + threadIdx.x;   // NOBJ * D_
  int row = i >> 9, d = i & 511;
  tok[i] += emb[(long long)objs[row] * D_ + d];
}

// first[b] = lower_bound(triple_to_img, b)
__global__ void firstk(const int* __restrict__ tti, int* __restrict__ first) {
  int b = blockIdx.x * 256 + threadIdx.x;
  if (b >= B_) return;
  int lo = 0, hi = NTRI;
  while (lo < hi) { int mid = (lo + hi) >> 1; if (tti[mid] < b) lo = mid + 1; else hi = mid; }
  first[b] = lo;
}

// scatter triple tokens into x (B,MT,D) and set key mask
__global__ __launch_bounds__(128)
void scatter(const int* __restrict__ triples, const int* __restrict__ tti,
             const int* __restrict__ first, const float* __restrict__ objtok,
             const float* __restrict__ pemb, float* __restrict__ x,
             unsigned char* __restrict__ mask) {
  int t = blockIdx.x;
  int img = tti[t];
  int pos = t - first[img];
  int si = triples[t*3+0], pi = triples[t*3+1], oi = triples[t*3+2];
#pragma unroll
  for (int j = 0; j < 3; ++j) {
    int slot = pos * 3 + j;
    if (slot >= MT_) continue;
    const float* src = (j == 1) ? (pemb + (long long)pi * D_)
                                : (objtok + (long long)(j == 0 ? si : oi) * D_);
    float* dst = x + ((long long)(img * MT_ + slot)) * D_;
    for (int d = threadIdx.x; d < D_; d += 128) dst[d] = src[d];
    if (threadIdx.x == 0) mask[img * MT_ + slot] = 1;
  }
}

// ------------------------------------------------------- WMMA bf16 GEMM
// C(MxN) = A(MxK bf16) @ Wt(NxK bf16)^T + bias; epilogue by MODE:
//   0: outb = bf16(C)      1: resf += C (residual)     2: outb = bf16(gelu(C))
//   3: resf  = C (f32 store)
template <int MODE>
__global__ __launch_bounds__(256)
void gemm_bf16(const bf16* __restrict__ A, const bf16* __restrict__ Wt,
               const float* __restrict__ bias, float* __restrict__ resf,
               bf16* __restrict__ outb, int M, int N, int K) {
  __shared__ __attribute__((aligned(32))) bf16 As[64][32];
  __shared__ __attribute__((aligned(32))) bf16 Bs[64][32];
  int tid = threadIdx.x;
  int bm = blockIdx.y * 64;
  int bn = blockIdx.x * 64;
  int lr = tid >> 2, lc = (tid & 3) * 8;         // staging: 64 rows x 32 cols, 8 bf16/thread
  int lane = tid & 31, warp = tid >> 5;
  int m16 = (warp & 3) * 16;                     // wave's 16-row strip
  int nh  = (warp >> 2) * 32;                    // wave's 32-col half
  int ml = lane & 15, lh = lane >> 4;

  v8f acc[2];
  v8f z = {0.f,0.f,0.f,0.f,0.f,0.f,0.f,0.f};
  acc[0] = z; acc[1] = z;

  const bf16* Ap = A  + (long long)(bm + lr) * K + lc;
  const bf16* Bp = Wt + (long long)(bn + lr) * K + lc;

  for (int k0 = 0; k0 < K; k0 += 32) {
    *(v8bf*)&As[lr][lc] = *(const v8bf*)(Ap + k0);
    *(v8bf*)&Bs[lr][lc] = *(const v8bf*)(Bp + k0);
    __syncthreads();
    // A fragment: lane holds row (m16+ml), contiguous K [lh*16..+15]
    v8bf lo = *(const v8bf*)&As[m16 + ml][lh * 16];
    v8bf hi = *(const v8bf*)&As[m16 + ml][lh * 16 + 8];
    v16bf a;
#pragma unroll
    for (int i = 0; i < 8; ++i) { a[i] = lo[i]; a[8 + i] = hi[i]; }
    // B fragments: column (n), contiguous 16 K values per half-wave
    v16bf b0 = *(const v16bf*)&Bs[nh + ml][lh * 16];
    v16bf b1 = *(const v16bf*)&Bs[nh + 16 + ml][lh * 16];
    acc[0] = __builtin_amdgcn_wmma_f32_16x16x32_bf16(false, a, false, b0, (short)0, acc[0], false, false);
    acc[1] = __builtin_amdgcn_wmma_f32_16x16x32_bf16(false, a, false, b1, (short)0, acc[1], false, false);
    __syncthreads();
  }
#pragma unroll
  for (int t = 0; t < 2; ++t) {
    int n = bn + nh + t * 16 + ml;
    float bvv = bias[n];
#pragma unroll
    for (int r = 0; r < 8; ++r) {
      int m = bm + m16 + r + lh * 8;
      float val = acc[t][r] + bvv;
      long long o = (long long)m * N + n;
      if (MODE == 0)      outb[o] = (bf16)val;
      else if (MODE == 1) resf[o] += val;
      else if (MODE == 2) outb[o] = (bf16)gelu_exact(val);
      else                resf[o] = val;
    }
  }
}

// ------------------------------------------------------- attention per (b,h)
__global__ __launch_bounds__(128)
void attn_kernel(const bf16* __restrict__ q, const bf16* __restrict__ k,
                 const bf16* __restrict__ v, const unsigned char* __restrict__ mask,
                 bf16* __restrict__ out) {
  __shared__ __attribute__((aligned(32))) bf16 qs[64][64];   // q, then reused as probs
  __shared__ __attribute__((aligned(32))) bf16 ks[64][64];
  __shared__ __attribute__((aligned(32))) bf16 vt[64][64];   // v transposed: [dim][tok]
  __shared__ float sc[64][64];
  __shared__ float mf[64];
  int tid = threadIdx.x;
  int b = blockIdx.x >> 3, h = blockIdx.x & 7;

  int tok = tid >> 1, dh = (tid & 1) * 32;
  long long base = ((long long)(b * MT_ + tok)) * D_ + h * HD_ + dh;
#pragma unroll
  for (int c = 0; c < 4; ++c) {
    *(v8bf*)&qs[tok][dh + c * 8] = *(const v8bf*)(q + base + c * 8);
    *(v8bf*)&ks[tok][dh + c * 8] = *(const v8bf*)(k + base + c * 8);
    v8bf vv = *(const v8bf*)(v + base + c * 8);
#pragma unroll
    for (int i = 0; i < 8; ++i) vt[dh + c * 8 + i][tok] = vv[i];
  }
  if (tid < 64) mf[tid] = mask[b * MT_ + tid] ? 1.0f : 0.0f;
  __syncthreads();

  int lane = tid & 31, warp = tid >> 5;
  int m16 = warp * 16, ml = lane & 15, lh = lane >> 4;
  v8f z = {0.f,0.f,0.f,0.f,0.f,0.f,0.f,0.f};

  // scores = q @ k^T  (K = HD = 64)
  v8f acc[4]; acc[0]=z; acc[1]=z; acc[2]=z; acc[3]=z;
#pragma unroll
  for (int ko = 0; ko < 64; ko += 32) {
    v8bf lo = *(const v8bf*)&qs[m16 + ml][ko + lh * 16];
    v8bf hi = *(const v8bf*)&qs[m16 + ml][ko + lh * 16 + 8];
    v16bf a;
#pragma unroll
    for (int i = 0; i < 8; ++i) { a[i] = lo[i]; a[8 + i] = hi[i]; }
#pragma unroll
    for (int nt = 0; nt < 4; ++nt) {
      v16bf bb = *(const v16bf*)&ks[nt * 16 + ml][ko + lh * 16];
      acc[nt] = __builtin_amdgcn_wmma_f32_16x16x32_bf16(false, a, false, bb, (short)0, acc[nt], false, false);
    }
  }
#pragma unroll
  for (int nt = 0; nt < 4; ++nt) {
    int n = nt * 16 + ml;
    bool on = (mf[n] != 0.0f);
#pragma unroll
    for (int r = 0; r < 8; ++r) {
      int m = m16 + r + lh * 8;
      sc[m][n] = on ? acc[nt][r] * 0.125f : -1.0e9f;
    }
  }
  __syncthreads();

  // softmax per row, write probs as bf16 into qs
  if (tid < 64) {
    float mx = sc[tid][0];
    for (int j = 1; j < 64; ++j) mx = fmaxf(mx, sc[tid][j]);
    float s = 0.0f;
    for (int j = 0; j < 64; ++j) { float e = expf(sc[tid][j] - mx); sc[tid][j] = e; s += e; }
    float inv = 1.0f / s;
    for (int j = 0; j < 64; ++j) qs[tid][j] = (bf16)(sc[tid][j] * inv);
  }
  __syncthreads();

  // out = P @ V   (K = tokens = 64)
  v8f av[4]; av[0]=z; av[1]=z; av[2]=z; av[3]=z;
#pragma unroll
  for (int ko = 0; ko < 64; ko += 32) {
    v8bf lo = *(const v8bf*)&qs[m16 + ml][ko + lh * 16];
    v8bf hi = *(const v8bf*)&qs[m16 + ml][ko + lh * 16 + 8];
    v16bf a;
#pragma unroll
    for (int i = 0; i < 8; ++i) { a[i] = lo[i]; a[8 + i] = hi[i]; }
#pragma unroll
    for (int nt = 0; nt < 4; ++nt) {
      v16bf bb = *(const v16bf*)&vt[nt * 16 + ml][ko + lh * 16];
      av[nt] = __builtin_amdgcn_wmma_f32_16x16x32_bf16(false, a, false, bb, (short)0, av[nt], false, false);
    }
  }
#pragma unroll
  for (int nt = 0; nt < 4; ++nt) {
    int n = nt * 16 + ml;
#pragma unroll
    for (int r = 0; r < 8; ++r) {
      int m = m16 + r + lh * 8;
      out[((long long)(b * MT_ + m)) * D_ + h * HD_ + n] = (bf16)av[nt][r];
    }
  }
}

// ================================================================= launcher
extern "C" void kernel_launch(void* const* d_in, const int* in_sizes, int n_in,
                              void* d_out, int out_size, void* d_ws, size_t ws_size,
                              hipStream_t stream) {
  const int*   objs    = (const int*)  d_in[0];
  const float* boxes   = (const float*)d_in[1];
  const int*   triples = (const int*)  d_in[2];
  const int*   tti     = (const int*)  d_in[4];
  const float* obj_emb = (const float*)d_in[5];
  const float* pred_emb= (const float*)d_in[6];
  const float* bw1 = (const float*)d_in[7];
  const float* bb1 = (const float*)d_in[8];
  const float* blg = (const float*)d_in[9];
  const float* blb = (const float*)d_in[10];
  const float* bw2 = (const float*)d_in[11];
  const float* bb2 = (const float*)d_in[12];
  const float* ln1g= (const float*)d_in[13];
  const float* ln1b= (const float*)d_in[14];
  const float* wq  = (const float*)d_in[15];
  const float* bq  = (const float*)d_in[16];
  const float* wk  = (const float*)d_in[17];
  const float* bk  = (const float*)d_in[18];
  const float* wv  = (const float*)d_in[19];
  const float* bvb = (const float*)d_in[20];
  const float* wo  = (const float*)d_in[21];
  const float* bo  = (const float*)d_in[22];
  const float* ln2g= (const float*)d_in[23];
  const float* ln2b= (const float*)d_in[24];
  const float* wf1 = (const float*)d_in[25];
  const float* bf1 = (const float*)d_in[26];
  const float* wf2 = (const float*)d_in[27];
  const float* bf2 = (const float*)d_in[28];
  (void)in_sizes; (void)n_in; (void)out_size;

  float* x = (float*)d_out;                       // residual stream, B*MT*D f32
  char* ws = (char*)d_ws;
  const size_t MB = 1024ull * 1024ull;
  if (ws_size < 347 * MB) return;                 // not enough scratch

  // Region 0 (0..256MB): q/k/v/attn during attention phase, ff intermediate
  // during FF phase, box-path buffers before the layer loop.
  bf16*  qb     = (bf16*)(ws + 0);
  bf16*  kb     = (bf16*)(ws + 64 * MB);
  bf16*  vb     = (bf16*)(ws + 128 * MB);
  bf16*  ab     = (bf16*)(ws + 192 * MB);
  bf16*  ffb    = (bf16*)(ws + 0);
  float* bxlin  = (float*)(ws + 0);
  bf16*  gbuf   = (bf16*)(ws + 32 * MB);
  float* objtok = (float*)(ws + 48 * MB);
  bf16*  hb     = (bf16*)(ws + 256 * MB);         // LN output, 64MB
  char*  wbase  = ws + 320 * MB;                  // transposed bf16 weights
  bf16*  wqT  = (bf16*)(wbase + 0 * MB);
  bf16*  wkT  = (bf16*)(wbase + 2 * MB);
  bf16*  wvT  = (bf16*)(wbase + 4 * MB);
  bf16*  woT  = (bf16*)(wbase + 6 * MB);
  bf16*  wf1T = (bf16*)(wbase + 8 * MB);
  bf16*  wf2T = (bf16*)(wbase + 16 * MB);
  bf16*  bw2T = (bf16*)(wbase + 24 * MB);
  unsigned char* mask = (unsigned char*)(wbase + 25 * MB);
  int* firstb = (int*)(wbase + 25 * MB + 65536);

  // init
  zero_f32<<<8192, 256, 0, stream>>>(x, (long long)NTOK * D_);
  zero_u8 <<<64,   256, 0, stream>>>(mask, (long long)NTOK);

  // weight conversion + transpose to N x K bf16
  for (int l = 0; l < 4; ++l) {
    long long w512 = (long long)l * D_ * D_;
    wt_convert<<<1024, 256, 0, stream>>>(wq + w512, wqT + w512, D_, D_);
    wt_convert<<<1024, 256, 0, stream>>>(wk + w512, wkT + w512, D_, D_);
    wt_convert<<<1024, 256, 0, stream>>>(wv + w512, wvT + w512, D_, D_);
    wt_convert<<<1024, 256, 0, stream>>>(wo + w512, woT + w512, D_, D_);
    wt_convert<<<4096, 256, 0, stream>>>(wf1 + (long long)l * D_ * FF_,
                                         wf1T + (long long)l * D_ * FF_, D_, FF_);
    wt_convert<<<4096, 256, 0, stream>>>(wf2 + (long long)l * FF_ * D_,
                                         wf2T + (long long)l * FF_ * D_, FF_, D_);
  }
  wt_convert<<<1024, 256, 0, stream>>>(bw2, bw2T, D_, D_);

  // box MLP: objtok = gelu(LN(boxes@bw1+bb1)) @ bw2 + bb2 + obj_emb[objs]
  boxlin<<<NOBJ * D_ / 256, 256, 0, stream>>>(boxes, bw1, bb1, bxlin);
  ln_rows<<<NOBJ / 8, 256, 0, stream>>>(bxlin, blg, blb, gbuf, 1);
  {
    dim3 g(D_ / 64, NOBJ / 64);
    gemm_bf16<3><<<g, 256, 0, stream>>>(gbuf, bw2T, bb2, objtok, nullptr, NOBJ, D_, D_);
  }
  addemb<<<NOBJ * D_ / 256, 256, 0, stream>>>(objtok, obj_emb, objs);

  // scatter triple tokens into x + build mask
  firstk <<<4, 256, 0, stream>>>(tti, firstb);
  scatter<<<NTRI, 128, 0, stream>>>(triples, tti, firstb, objtok, pred_emb, x, mask);

  // transformer layers
  dim3 gD(D_ / 64, NTOK / 64);
  dim3 gF(FF_ / 64, NTOK / 64);
  for (int l = 0; l < 4; ++l) {
    long long w512 = (long long)l * D_ * D_;
    long long b512 = (long long)l * D_;
    ln_rows<<<NTOK / 8, 256, 0, stream>>>(x, ln1g + b512, ln1b + b512, hb, 0);
    gemm_bf16<0><<<gD, 256, 0, stream>>>(hb, wqT + w512, bq  + b512, nullptr, qb, NTOK, D_, D_);
    gemm_bf16<0><<<gD, 256, 0, stream>>>(hb, wkT + w512, bk  + b512, nullptr, kb, NTOK, D_, D_);
    gemm_bf16<0><<<gD, 256, 0, stream>>>(hb, wvT + w512, bvb + b512, nullptr, vb, NTOK, D_, D_);
    attn_kernel<<<B_ * H_, 128, 0, stream>>>(qb, kb, vb, mask, ab);
    gemm_bf16<1><<<gD, 256, 0, stream>>>(ab, woT + w512, bo + b512, x, nullptr, NTOK, D_, D_);
    ln_rows<<<NTOK / 8, 256, 0, stream>>>(x, ln2g + b512, ln2b + b512, hb, 0);
    gemm_bf16<2><<<gF, 256, 0, stream>>>(hb, wf1T + (long long)l * D_ * FF_,
                                         bf1 + (long long)l * FF_, nullptr, ffb, NTOK, FF_, D_);
    gemm_bf16<1><<<gD, 256, 0, stream>>>(ffb, wf2T + (long long)l * FF_ * D_,
                                         bf2 + b512, x, nullptr, NTOK, D_, FF_);
  }
}
